// Attention_Aggregator_89232240541864
// MI455X (gfx1250) — compile-verified, matrix operand order checked by reading, not run on hardware
//
#include <hip/hip_runtime.h>

#define BATCH  64
#define SEQ    2048
#define DMODEL 512
#define ATT    256

typedef __attribute__((ext_vector_type(16))) __bf16 v16bf;
typedef __attribute__((ext_vector_type(8)))  __bf16 v8bf;
typedef __attribute__((ext_vector_type(8)))  float  v8f;

#define LDS_ROWS  32
#define LDS_PITCH (DMODEL + 8)   // bf16 elems; keeps 16B alignment, staggers banks (+4 banks/row)

// ---------------------------------------------------------------------------
// Kernel 1: scores[b,s] = ae . tanh( input[b,s,:] @ Wa^T + ba )
// One wave = 16 sequence rows (M).  N = 256 attention units in 16x16 tiles.
// K = 512 model dims in 16 steps of 32 (bf16 WMMA).
// ---------------------------------------------------------------------------
__global__ __launch_bounds__(256)
void attn_scores_kernel(const float* __restrict__ input,
                        const float* __restrict__ Wa,
                        const float* __restrict__ ba,
                        const float* __restrict__ ae,
                        float* __restrict__ scores)
{
  __shared__ __align__(16) __bf16 WaLds[LDS_ROWS][LDS_PITCH];

  const int tid  = threadIdx.x;
  const int wave = tid >> 5;
  const int lane = tid & 31;
  const int lo16 = lane & 15;
  const int hi   = lane >> 4;

  const int b  = blockIdx.y;
  const int s0 = blockIdx.x * 128 + wave * 16;

  // ---- Preload A fragments for all 16 K-steps (ISA 16-bit A 16x32 layout:
  //      lane<16 -> M=lane, K chunks {0..7,16..23}; lane>=16 -> K {8..15,24..31}).
  const float* arow = input + ((size_t)b * SEQ + (size_t)(s0 + lo16)) * DMODEL;
  v16bf afrag[16];
  #pragma unroll
  for (int k = 0; k < 16; ++k) {
    const int col = k * 32 + hi * 8;
    const float4 f0 = *(const float4*)(arow + col);
    const float4 f1 = *(const float4*)(arow + col + 4);
    const float4 g0 = *(const float4*)(arow + col + 16);
    const float4 g1 = *(const float4*)(arow + col + 20);
    v16bf a;
    a[0]  = (__bf16)f0.x; a[1]  = (__bf16)f0.y; a[2]  = (__bf16)f0.z; a[3]  = (__bf16)f0.w;
    a[4]  = (__bf16)f1.x; a[5]  = (__bf16)f1.y; a[6]  = (__bf16)f1.z; a[7]  = (__bf16)f1.w;
    a[8]  = (__bf16)g0.x; a[9]  = (__bf16)g0.y; a[10] = (__bf16)g0.z; a[11] = (__bf16)g0.w;
    a[12] = (__bf16)g1.x; a[13] = (__bf16)g1.y; a[14] = (__bf16)g1.z; a[15] = (__bf16)g1.w;
    afrag[k] = a;
  }

  float partial[8] = {0.f,0.f,0.f,0.f,0.f,0.f,0.f,0.f};

  for (int nb = 0; nb < ATT / LDS_ROWS; ++nb) {      // 8 N-blocks of 32 rows
    __syncthreads();
    // Stage 32 rows of Wa (f32 -> bf16) into LDS, coalesced float4 loads.
    for (int i = tid; i < LDS_ROWS * (DMODEL / 4); i += 256) {
      const int r = i >> 7;              // row within block
      const int c = (i & 127) << 2;      // float column
      const float4 v = *(const float4*)(Wa + ((size_t)(nb * LDS_ROWS + r)) * DMODEL + c);
      WaLds[r][c + 0] = (__bf16)v.x;
      WaLds[r][c + 1] = (__bf16)v.y;
      WaLds[r][c + 2] = (__bf16)v.z;
      WaLds[r][c + 3] = (__bf16)v.w;
    }
    __syncthreads();

    #pragma unroll
    for (int nt = 0; nt < 2; ++nt) {                 // 2 N-tiles per block
      v8f acc = {};
      // B fragment: lane = N column (Wa row), contiguous K per lane -> no transpose.
      const __bf16* brow = &WaLds[nt * 16 + lo16][0];
      #pragma unroll
      for (int k = 0; k < 16; ++k) {
        const int kb = k * 32 + hi * 8;
        const v8bf blo = *(const v8bf*)(brow + kb);
        const v8bf bhi = *(const v8bf*)(brow + kb + 16);
        const v16bf bfrag = __builtin_shufflevector(
            blo, bhi, 0,1,2,3,4,5,6,7,8,9,10,11,12,13,14,15);
        acc = __builtin_amdgcn_wmma_f32_16x16x32_bf16(
            false, afrag[k], false, bfrag, (short)0, acc, false, false);
      }
      // Fold: partial[r] over n = nb*32 + nt*16 + lane-N
      const int   n   = nb * 32 + nt * 16 + lo16;
      const float aev = ae[n];
      const float bav = ba[n];
      #pragma unroll
      for (int r = 0; r < 8; ++r)
        partial[r] += aev * tanhf(acc[r] + bav);
    }
  }

  // Reduce over the 16 N-lanes within each half-wave (C/D layout: VGPR r holds
  // M=r in lanes 0-15 and M=r+8 in lanes 16-31).
  #pragma unroll
  for (int r = 0; r < 8; ++r) {
    float v = partial[r];
    v += __shfl_xor(v, 1, 32);
    v += __shfl_xor(v, 2, 32);
    v += __shfl_xor(v, 4, 32);
    v += __shfl_xor(v, 8, 32);
    partial[r] = v;
  }
  if (lo16 == 0) {
    float* srow = scores + (size_t)b * SEQ + s0;
    #pragma unroll
    for (int r = 0; r < 8; ++r)
      srow[r + 8 * hi] = partial[r];
  }
}

// ---------------------------------------------------------------------------
// Kernel 2: softmax over the sequence dim, one block per batch row.
// ---------------------------------------------------------------------------
__global__ __launch_bounds__(256)
void attn_softmax_kernel(const float* __restrict__ scores,
                         float* __restrict__ out_w)
{
  const int b   = blockIdx.x;
  const int tid = threadIdx.x;
  const float* srow = scores + (size_t)b * SEQ;

  __shared__ float red[256];

  float v[8];
  float mx = -3.402823466e38f;
  #pragma unroll
  for (int i = 0; i < 8; ++i) {
    v[i] = srow[tid + i * 256];
    mx = fmaxf(mx, v[i]);
  }
  red[tid] = mx;
  __syncthreads();
  for (int off = 128; off > 0; off >>= 1) {
    if (tid < off) red[tid] = fmaxf(red[tid], red[tid + off]);
    __syncthreads();
  }
  mx = red[0];
  __syncthreads();

  float sum = 0.f;
  #pragma unroll
  for (int i = 0; i < 8; ++i) {
    v[i] = __expf(v[i] - mx);
    sum += v[i];
  }
  red[tid] = sum;
  __syncthreads();
  for (int off = 128; off > 0; off >>= 1) {
    if (tid < off) red[tid] += red[tid + off];
    __syncthreads();
  }
  const float inv = 1.f / red[0];

  float* wrow = out_w + (size_t)b * SEQ;
  #pragma unroll
  for (int i = 0; i < 8; ++i)
    wrow[tid + i * 256] = v[i] * inv;
}

// ---------------------------------------------------------------------------
// Kernel 3: out[b,f] = sum_s input[b,s,f] * w[b,s].  grid=(4 f-chunks, 64 b),
// block=128 threads, weights cached in LDS, no atomics, single input pass.
// ---------------------------------------------------------------------------
__global__ __launch_bounds__(128)
void attn_pool_kernel(const float* __restrict__ input,
                      const float* __restrict__ w,
                      float* __restrict__ out)
{
  const int b   = blockIdx.y;
  const int tid = threadIdx.x;
  const int f   = blockIdx.x * 128 + tid;

  __shared__ float wl[SEQ];           // 8 KB
  for (int s = tid; s < SEQ; s += 128)
    wl[s] = w[(size_t)b * SEQ + s];
  __syncthreads();

  const float* col = input + (size_t)b * SEQ * DMODEL + f;
  float acc = 0.f;
  for (int s = 0; s < SEQ; s += 8) {
    __builtin_prefetch((const void*)(col + (size_t)(s + 64) * DMODEL), 0, 0);
    #pragma unroll
    for (int j = 0; j < 8; ++j)
      acc = fmaf(col[(size_t)(s + j) * DMODEL], wl[s + j], acc);
  }
  out[(size_t)b * DMODEL + f] = acc;
}

// ---------------------------------------------------------------------------
extern "C" void kernel_launch(void* const* d_in, const int* in_sizes, int n_in,
                              void* d_out, int out_size, void* d_ws, size_t ws_size,
                              hipStream_t stream) {
  const float* input = (const float*)d_in[0];  // [64,2048,512]
  const float* Wa    = (const float*)d_in[1];  // [256,512]
  const float* ba    = (const float*)d_in[2];  // [256]
  const float* ae    = (const float*)d_in[3];  // [256]

  float* out_feat = (float*)d_out;                       // [64,512]
  float* out_w    = out_feat + (size_t)BATCH * DMODEL;   // [64,2048]
  float* scores   = (float*)d_ws;                        // [64,2048] scratch (512 KB)

  attn_scores_kernel<<<dim3(SEQ / 128, BATCH), 256, 0, stream>>>(
      input, Wa, ba, ae, scores);
  attn_softmax_kernel<<<BATCH, 256, 0, stream>>>(scores, out_w);
  attn_pool_kernel<<<dim3(DMODEL / 128, BATCH), 128, 0, stream>>>(
      input, out_w, out_feat);
}
